// AdaptiveAttention_64622077936012
// MI455X (gfx1250) — matrix operand order repaired and checked by reading, not compiled, not run on hardware
//
#include <hip/hip_runtime.h>
#include <hip/hip_bf16.h>

typedef __attribute__((ext_vector_type(16))) _Float16 v16h;
typedef __attribute__((ext_vector_type(8)))  _Float16 h8;
typedef __attribute__((ext_vector_type(8)))  float    v8f;
typedef __attribute__((ext_vector_type(4)))  float    f4;
typedef __attribute__((ext_vector_type(4)))  unsigned int u4;
typedef __attribute__((ext_vector_type(4)))  int      i4;

static constexpr int kB  = 4;
static constexpr int kS  = 2048;
static constexpr int kD  = 1024;
static constexpr int kH  = 16;
static constexpr int kDK = 64;
static constexpr int kBS = kB * kS;   // 8192

union U16h { v16h v; h8 h2[2]; _Float16 e[16]; };

#if __has_builtin(__builtin_amdgcn_global_load_async_to_lds_b128)
#define HAVE_ASYNC_LDS 1
#else
#define HAVE_ASYNC_LDS 0
#endif

// 16-byte global -> LDS copy; async DMA (ASYNCcnt) on CDNA5 when available.
// Builtin takes (v4i32 AS1* src, v4i32 AS3* dst, imm offset, imm cpol).
// Pointers are materialized via integer casts: global generic == AS(1) value,
// and generic LDS pointers' low 32 bits are the LDS offset (ISA 10.2).
template <typename T>
__device__ __forceinline__ void copy16_g2l(const T* g, T* l) {
#if HAVE_ASYNC_LDS
  __builtin_amdgcn_global_load_async_to_lds_b128(
      (__attribute__((address_space(1))) i4*)(unsigned long long)g,
      (__attribute__((address_space(3))) i4*)(unsigned int)(unsigned long long)l,
      0, 0);
#else
  *(u4*)l = *(const u4*)g;
#endif
}

__device__ __forceinline__ void wait_async_lds() {
#if HAVE_ASYNC_LDS
#if __has_builtin(__builtin_amdgcn_s_wait_asynccnt)
  __builtin_amdgcn_s_wait_asynccnt(0);
#else
  asm volatile("s_wait_asynccnt 0x0" ::: "memory");
#endif
#endif
}

__device__ __forceinline__ v8f wmma_f16(v16h a, v16h b, v8f c) {
  // D = A(16x32 f16) * B(32x16 f16) + C(16x16 f32)
  return __builtin_amdgcn_wmma_f32_16x16x32_f16(false, a, false, b, (short)0, c,
                                                false, false);
}

// 16 contiguous halves (32B) -> B-fragment (halves h = K kb2+h, col = lane&15)
__device__ __forceinline__ v16h load16h(const _Float16* p) {
  U16h u;
  u.h2[0] = *(const h8*)(p);
  u.h2[1] = *(const h8*)(p + 8);
  return u.v;
}

// ---------------------------------------------------------------------------
// Generic GEMM: C[M,N] = A[M,K] * W[K,N] + bias.
// Block tile 256x64, BK=32; 8 waves, each computes a 32x64 strip = 8 WMMA/step.
// A tile staged raw (async DMA to LDS), converted to f16 during fragment build.
// W tile staged transposed as f16 so B-fragments are contiguous ds_load_b128.
// MODE 0: outH = f16((acc+bias)*scale)
// MODE 2: outH = f16(gateSrc * sigmoid(acc+bias))   (context gating)
// MODE 3: outF = acc + bias + resid                 (output proj + residual)
// ---------------------------------------------------------------------------
template <typename AT, int MODE>
__global__ __launch_bounds__(256) void gemm_kernel(
    const AT* __restrict__ A, const float* __restrict__ W,
    const float* __restrict__ bias, _Float16* __restrict__ outH,
    float* __restrict__ outF, const float* __restrict__ gateSrc,
    const float* __restrict__ resid, float scale, int M, int N, int K) {
  __shared__ __align__(16) AT       AsT[256 * 32];  // raw A tile (f32:32KB f16:16KB)
  __shared__ __align__(16) _Float16 Bs[64 * 40];    // transposed [n][k] f16, padded

  const int tid  = threadIdx.x;
  const int lane = tid & 31;
  const int wid  = tid >> 5;
  const int m15  = lane & 15;
  const int hi   = lane >> 4;
  const int bm   = blockIdx.y * 256;
  const int bn   = blockIdx.x * 64;

  v8f acc[2][4] = {};

  constexpr int EPC = 16 / (int)sizeof(AT);  // elements per 16B copy
  constexpr int SPR = 32 / EPC;              // 16B slots per 32-wide row
  constexpr int NC  = (256 * 32) / (EPC * 256);

  for (int k0 = 0; k0 < K; k0 += 32) {
    // ---- stage A tile (256x32, raw) via async-to-LDS DMA ----
#pragma unroll
    for (int c = 0; c < NC; ++c) {
      int l   = tid + c * 256;
      int row = l / SPR;
      int kq  = (l % SPR) * EPC;
      copy16_g2l(A + (size_t)(bm + row) * K + k0 + kq, &AsT[row * 32 + kq]);
    }
    // ---- stage W tile (32x64) transposed into LDS as f16 ----
#pragma unroll
    for (int j = 0; j < 2; ++j) {
      int l    = tid + j * 256;           // 512 float4 slots
      int krow = l >> 4;
      int nq   = (l & 15) << 2;
      f4 v = *(const f4*)(W + (size_t)(k0 + krow) * N + bn + nq);
#pragma unroll
      for (int ii = 0; ii < 4; ++ii) Bs[(nq + ii) * 40 + krow] = (_Float16)v[ii];
    }
    if (k0 + 32 < K)  // CDNA5 global_prefetch_b8 for next W tile
      __builtin_prefetch(W + (size_t)(k0 + 32 + (tid >> 4)) * N + bn + ((tid & 15) << 2), 0, 1);
    wait_async_lds();
    __syncthreads();

    // ---- B fragments, shared by both M-halves of this wave ----
    v16h bf[4];
#pragma unroll
    for (int nt = 0; nt < 4; ++nt)
      bf[nt] = load16h(&Bs[(nt * 16 + m15) * 40 + hi * 16]);

#pragma unroll
    for (int half = 0; half < 2; ++half) {
      const AT* arow = &AsT[(wid * 32 + half * 16 + m15) * 32 + hi * 16];
      U16h u;
      if constexpr (sizeof(AT) == 4) {
        f4 a0 = *(const f4*)(arow),      a1 = *(const f4*)(arow + 4);
        f4 a2 = *(const f4*)(arow + 8), a3 = *(const f4*)(arow + 12);
#pragma unroll
        for (int j = 0; j < 4; ++j) {
          u.e[j]      = (_Float16)a0[j];
          u.e[4 + j]  = (_Float16)a1[j];
          u.e[8 + j]  = (_Float16)a2[j];
          u.e[12 + j] = (_Float16)a3[j];
        }
      } else {
        u.h2[0] = *(const h8*)(arow);
        u.h2[1] = *(const h8*)(arow + 8);
      }
#pragma unroll
      for (int nt = 0; nt < 4; ++nt)
        acc[half][nt] = wmma_f16(u.v, bf[nt], acc[half][nt]);
    }
    __syncthreads();
  }

  // ---- epilogue: C element (r) -> row = hi*8+r, col = lane&15 ----
#pragma unroll
  for (int half = 0; half < 2; ++half) {
    const int mlocal = wid * 32 + half * 16 + hi * 8;
#pragma unroll
    for (int nt = 0; nt < 4; ++nt) {
      int gcol   = bn + nt * 16 + m15;
      float bval = bias[gcol];
#pragma unroll
      for (int r = 0; r < 8; ++r) {
        int grow   = bm + mlocal + r;
        size_t idx = (size_t)grow * N + gcol;
        float x    = acc[half][nt][r] + bval;
        if constexpr (MODE == 0) {
          outH[idx] = (_Float16)(x * scale);
        } else if constexpr (MODE == 2) {
          float g   = 1.0f / (1.0f + __expf(-x));
          outH[idx] = (_Float16)(gateSrc[idx] * g);
        } else {
          outF[idx] = x + resid[idx];
        }
      }
    }
  }
}

// ---------------------------------------------------------------------------
// scores[b,h,q,k] = (Q/8) . K  via register-direct WMMA, K-dim = DK = 64
// ---------------------------------------------------------------------------
__global__ __launch_bounds__(256) void scores_kernel(
    const _Float16* __restrict__ Qh, const _Float16* __restrict__ Kh,
    float* __restrict__ attn) {
  const int lane = threadIdx.x & 31, wid = threadIdx.x >> 5;
  const int m15 = lane & 15, hi = lane >> 4;
  const int kt = blockIdx.x * 8 + wid;  // 0..127
  const int qt = blockIdx.y;            // 0..127
  const int bh = blockIdx.z;            // 0..63
  const int b = bh >> 4, h = bh & 15;

  const size_t qbase = ((size_t)b * kS + qt * 16 + m15) * kD + h * kDK + hi * 16;
  const size_t kbase = ((size_t)b * kS + kt * 16 + m15) * kD + h * kDK + hi * 16;

  v8f acc = {};
#pragma unroll
  for (int ks = 0; ks < 2; ++ks) {  // dk 0..31, 32..63
    U16h a, bf;
    a.h2[0]  = *(const h8*)(Qh + qbase + ks * 32);
    a.h2[1]  = *(const h8*)(Qh + qbase + ks * 32 + 8);
    bf.h2[0] = *(const h8*)(Kh + kbase + ks * 32);
    bf.h2[1] = *(const h8*)(Kh + kbase + ks * 32 + 8);
    acc = wmma_f16(a.v, bf.v, acc);
  }
  float* out = attn + ((size_t)bh * kS + qt * 16) * kS + kt * 16;
#pragma unroll
  for (int r = 0; r < 8; ++r) out[(size_t)(hi * 8 + r) * kS + m15] = acc[r];
}

// ---------------------------------------------------------------------------
// row-wise softmax over 2048, scaled by per-(b,h) sigmoid gate, in place
// ---------------------------------------------------------------------------
__global__ __launch_bounds__(256) void softmax_kernel(float* __restrict__ attn,
                                                      const float* __restrict__ gate) {
  __shared__ float red[256];
  const size_t rowId = blockIdx.x;          // 0 .. B*H*S-1
  const int bh = (int)(rowId / kS);
  float* row = attn + rowId * (size_t)kS;
  const int tid = threadIdx.x;

  float vals[8];
  float mx = -1e30f;
#pragma unroll
  for (int j = 0; j < 8; ++j) {
    vals[j] = row[tid + j * 256];
    mx = fmaxf(mx, vals[j]);
  }
  red[tid] = mx; __syncthreads();
  for (int s = 128; s > 0; s >>= 1) {
    if (tid < s) red[tid] = fmaxf(red[tid], red[tid + s]);
    __syncthreads();
  }
  mx = red[0]; __syncthreads();
  float sum = 0.f;
#pragma unroll
  for (int j = 0; j < 8; ++j) { vals[j] = __expf(vals[j] - mx); sum += vals[j]; }
  red[tid] = sum; __syncthreads();
  for (int s = 128; s > 0; s >>= 1) {
    if (tid < s) red[tid] += red[tid + s];
    __syncthreads();
  }
  float g = gate[bh] / red[0];
#pragma unroll
  for (int j = 0; j < 8; ++j) row[tid + j * 256] = vals[j] * g;
}

// ---------------------------------------------------------------------------
// ctx[b,q,h*64+dk] = attn[b,h,q,:] @ V[b,:,h*64+dk]   (K = 2048, WMMA)
// VT layout: [b,h,dk,s] f16 so B fragments are 32B-contiguous loads
// ---------------------------------------------------------------------------
__global__ __launch_bounds__(256) void ctx_kernel(const float* __restrict__ attn,
                                                  const _Float16* __restrict__ VT,
                                                  float* __restrict__ ctx) {
  const int lane = threadIdx.x & 31, wid = threadIdx.x >> 5;
  const int m15 = lane & 15, hi = lane >> 4;
  const int bh = blockIdx.z;
  const int b = bh >> 4, h = bh & 15;
  const int rowBase = blockIdx.y * 128 + wid * 16;
  const size_t arow  = ((size_t)bh * kS + rowBase + m15) * kS;
  const size_t vbase = (size_t)bh * kDK * kS;

  v8f acc[4] = {};
  for (int k0 = 0; k0 < kS; k0 += 32) {
    const float* ap = attn + arow + k0 + hi * 16;
    f4 v0 = *(const f4*)(ap),      v1 = *(const f4*)(ap + 4);
    f4 v2 = *(const f4*)(ap + 8), v3 = *(const f4*)(ap + 12);
    U16h a;
#pragma unroll
    for (int j = 0; j < 4; ++j) {
      a.e[j]      = (_Float16)v0[j];
      a.e[4 + j]  = (_Float16)v1[j];
      a.e[8 + j]  = (_Float16)v2[j];
      a.e[12 + j] = (_Float16)v3[j];
    }
#pragma unroll
    for (int nt = 0; nt < 4; ++nt) {
      const _Float16* vp = VT + vbase + (size_t)(nt * 16 + m15) * kS + k0 + hi * 16;
      acc[nt] = wmma_f16(a.v, load16h(vp), acc[nt]);
    }
  }
#pragma unroll
  for (int nt = 0; nt < 4; ++nt)
#pragma unroll
    for (int r = 0; r < 8; ++r) {
      int row = rowBase + hi * 8 + r;
      int dk  = nt * 16 + m15;
      ctx[((size_t)b * kS + row) * kD + h * kDK + dk] = acc[nt][r];
    }
}

__global__ void transpose_v_kernel(const _Float16* __restrict__ Vh,
                                   _Float16* __restrict__ VT) {
  size_t i = (size_t)blockIdx.x * blockDim.x + threadIdx.x;  // over B*S*D
  int d = (int)(i & (kD - 1));
  size_t bs = i >> 10;
  int s = (int)(bs & (kS - 1));
  int b = (int)(bs >> 11);
  int h = d >> 6, dk = d & 63;
  VT[(((size_t)(b * kH + h)) * kDK + dk) * kS + s] = Vh[i];
}

__global__ void mean_pool_kernel(const float* __restrict__ q,
                                 float* __restrict__ qmean) {
  int i = blockIdx.x * blockDim.x + threadIdx.x;  // B*D
  int b = i >> 10, d = i & (kD - 1);
  const float* p = q + (size_t)b * kS * kD + d;
  float s = 0.f;
  for (int t = 0; t < kS; ++t) s += p[(size_t)t * kD];
  qmean[(size_t)b * kD + d] = s * (1.0f / kS);
}

__global__ void gate_kernel(const float* __restrict__ qmean,
                            const float* __restrict__ Wg,
                            const float* __restrict__ bg,
                            float* __restrict__ gate) {
  int i = threadIdx.x;  // 64 threads = B*H
  int b = i >> 4, h = i & 15;
  float s = bg[h];
  for (int d = 0; d < kD; ++d) s += qmean[b * kD + d] * Wg[d * kH + h];
  gate[i] = 1.0f / (1.0f + __expf(-s));
}

__global__ __launch_bounds__(256) void ln_kernel(const float* __restrict__ x,
                                                 const float* __restrict__ gamma,
                                                 const float* __restrict__ beta,
                                                 float* __restrict__ out) {
  __shared__ float red[256];
  const size_t row = blockIdx.x;
  const float* xr = x + row * kD;
  const int tid = threadIdx.x;
  f4 v = *(const f4*)(xr + tid * 4);
  red[tid] = v[0] + v[1] + v[2] + v[3];
  __syncthreads();
  for (int s = 128; s > 0; s >>= 1) {
    if (tid < s) red[tid] += red[tid + s];
    __syncthreads();
  }
  float mu = red[0] * (1.0f / kD);
  __syncthreads();
  float vs = 0.f;
#pragma unroll
  for (int j = 0; j < 4; ++j) { float d = v[j] - mu; vs += d * d; }
  red[tid] = vs; __syncthreads();
  for (int s = 128; s > 0; s >>= 1) {
    if (tid < s) red[tid] += red[tid + s];
    __syncthreads();
  }
  float inv = rsqrtf(red[0] * (1.0f / kD) + 1e-5f);
  float* o = out + row * kD;
#pragma unroll
  for (int j = 0; j < 4; ++j) {
    int c = tid * 4 + j;
    o[c] = (v[j] - mu) * inv * gamma[c] + beta[c];
  }
}

extern "C" void kernel_launch(void* const* d_in, const int* in_sizes, int n_in,
                              void* d_out, int out_size, void* d_ws, size_t ws_size,
                              hipStream_t stream) {
  (void)in_sizes; (void)n_in; (void)out_size; (void)ws_size;
  const float* query = (const float*)d_in[0];
  const float* key   = (const float*)d_in[1];
  const float* value = (const float*)d_in[2];
  const float* Wq = (const float*)d_in[3];  const float* bq = (const float*)d_in[4];
  const float* Wk = (const float*)d_in[5];  const float* bk = (const float*)d_in[6];
  const float* Wv = (const float*)d_in[7];  const float* bv = (const float*)d_in[8];
  const float* Wo = (const float*)d_in[9];  const float* bo = (const float*)d_in[10];
  const float* Wg = (const float*)d_in[11]; const float* bg = (const float*)d_in[12];
  const float* Wc = (const float*)d_in[13]; const float* bc = (const float*)d_in[14];
  const float* gamma = (const float*)d_in[15];
  const float* beta  = (const float*)d_in[16];

  float* out  = (float*)d_out;
  float* attn = out + (size_t)kBS * kD;   // [B,H,S,S] region of d_out

  char* ws = (char*)d_ws;
  size_t off = 0;
  auto take = [&](size_t bytes) -> void* {
    void* p = ws + off;
    off += (bytes + 255) & ~(size_t)255;
    return p;
  };
  _Float16* Qh  = (_Float16*)take((size_t)kBS * kD * 2);
  _Float16* Kh  = (_Float16*)take((size_t)kBS * kD * 2);
  _Float16* Vh  = (_Float16*)take((size_t)kBS * kD * 2);  // later reused as gated
  _Float16* VTh = (_Float16*)take((size_t)kBS * kD * 2);
  float*    ctx = (float*)take((size_t)kBS * kD * 4);     // later reused as x
  float*  qmean = (float*)take((size_t)kB * kD * 4);
  float*   gate = (float*)take((size_t)kB * kH * 4);
  _Float16* gatedH = Vh;   // V consumed by transpose before cgate GEMM runs
  float*    xF     = ctx;  // ctx consumed by cgate GEMM before Wo GEMM writes

  mean_pool_kernel<<<(kB * kD) / 256, 256, 0, stream>>>(query, qmean);
  gate_kernel<<<1, 64, 0, stream>>>(qmean, Wg, bg, gate);

  dim3 gg(kD / 64, kBS / 256);  // (16, 32)
  gemm_kernel<float, 0><<<gg, 256, 0, stream>>>(query, Wq, bq, Qh, nullptr,
                                                nullptr, nullptr, 0.125f, kBS, kD, kD);
  gemm_kernel<float, 0><<<gg, 256, 0, stream>>>(key, Wk, bk, Kh, nullptr,
                                                nullptr, nullptr, 1.0f, kBS, kD, kD);
  gemm_kernel<float, 0><<<gg, 256, 0, stream>>>(value, Wv, bv, Vh, nullptr,
                                                nullptr, nullptr, 1.0f, kBS, kD, kD);
  transpose_v_kernel<<<(kBS * kD) / 256, 256, 0, stream>>>(Vh, VTh);

  scores_kernel<<<dim3(16, 128, 64), 256, 0, stream>>>(Qh, Kh, attn);
  softmax_kernel<<<kB * kH * kS, 256, 0, stream>>>(attn, gate);
  ctx_kernel<<<dim3(1, 16, 64), 256, 0, stream>>>(attn, VTh, ctx);

  gemm_kernel<float, 2><<<gg, 256, 0, stream>>>(ctx, Wc, bc, gatedH, nullptr,
                                                ctx, nullptr, 1.0f, kBS, kD, kD);
  gemm_kernel<_Float16, 3><<<gg, 256, 0, stream>>>(gatedH, Wo, bo, nullptr, xF,
                                                   nullptr, query, 1.0f, kBS, kD, kD);
  ln_kernel<<<kBS, 256, 0, stream>>>(xF, gamma, beta, out);
}
